// LSTMModel_1382979469640
// MI455X (gfx1250) — compile-verified, matrix operand order checked by reading, not run on hardware
//
#include <hip/hip_runtime.h>
#include <hip/hip_bf16.h>

// ---------------------------------------------------------------------------
// 2-layer LSTM scan on gfx1250 (MI455X), bf16 WMMA + persistent-kernel scan.
//
// Shapes: B=128, T=512, D=256, H=512, 4H=2048.
//   xp   = x @ proj_w^T + proj_b                       (bf16, precomputed, [T][B][D])
//   step: gates0 = [xp_t | h0] @ [wx0|wh0]^T + b0sum   (K = 256+512)
//         cell -> h0,c0
//         gates1 = [h0 | h1] @ [wx1|wh1]^T + b1sum     (K = 512+512)
//         cell -> h1,c1
//   out  = relu(h1 @ fc1^T + b) @ fc2^T + b2
//
// Persistent kernel: 32 WGs (one per h-block hb) x 8 waves (one per batch
// block mb). Each block stages its 64 weight columns (4 gates x 16) of all
// four matrices into LDS ONCE (224 KB, async global->LDS), then runs all 512
// timesteps reading B fragments from LDS (ds_load_b128) and A fragments
// (xp_t / h state) from L2. Device-scope epoch barrier between layer phases.
// ---------------------------------------------------------------------------

typedef __bf16 bf16;
typedef __attribute__((ext_vector_type(8)))  bf16  v8bf;
typedef __attribute__((ext_vector_type(16))) bf16  v16bf;
typedef __attribute__((ext_vector_type(8)))  float v8f;
typedef int v4i __attribute__((vector_size(16)));

static constexpr int kB  = 128;
static constexpr int kT  = 512;
static constexpr int kD  = 256;
static constexpr int kH  = 512;
static constexpr int kG  = 4 * kH;       // 2048 gate columns
static constexpr int kBH = kB * kH;      // 65536 state elements
static constexpr int NWG = 32;           // persistent workgroups (= h blocks)

// ---------------------------------------------------------------------------
// Fragment helpers (documented CDNA5 16-bit A layout; B mirrored; C/D layout
// VGPR r -> M = r + 8*(lane>=16), N = lane&15).
// ---------------------------------------------------------------------------
__device__ __forceinline__ v16bf make_frag(v8bf lo, v8bf hi) {
  return __builtin_shufflevector(lo, hi, 0, 1, 2, 3, 4, 5, 6, 7,
                                         8, 9, 10, 11, 12, 13, 14, 15);
}

__device__ __forceinline__ v16bf load_frag_bf16(const bf16* __restrict__ base,
                                                int ld, int row0, int k0, int lane) {
  const int r  = row0 + (lane & 15);
  const int kb = k0 + ((lane >> 4) << 3);
  const bf16* p = base + (size_t)r * ld + kb;
  return make_frag(*(const v8bf*)(p), *(const v8bf*)(p + 16));
}

// B fragment from an LDS-staged weight slice laid out as [64 rows][K] where
// row = gate*16 + col-in-tile.
__device__ __forceinline__ v16bf load_frag_lds(const bf16* base, int ldK,
                                               int gate, int k0, int lane) {
  const bf16* p = base + (gate * 16 + (lane & 15)) * ldK + k0 + ((lane >> 4) << 3);
  return make_frag(*(const v8bf*)(p), *(const v8bf*)(p + 16));
}

// A fragment straight from f32 memory (converted in-register -> no x_bf16 buf)
__device__ __forceinline__ v16bf load_frag_f32(const float* __restrict__ base,
                                               int ld, int row0, int k0, int lane) {
  const int r  = row0 + (lane & 15);
  const int kb = k0 + ((lane >> 4) << 3);
  const float* p = base + (size_t)r * ld + kb;
  v16bf a;
#pragma unroll
  for (int i = 0; i < 8; ++i) {
    a[i]     = (bf16)p[i];
    a[8 + i] = (bf16)p[i + 16];
  }
  return a;
}

__device__ __forceinline__ v8f wmma_bf16(v16bf a, v16bf b, v8f c) {
  return __builtin_amdgcn_wmma_f32_16x16x32_bf16(
      /*neg_a=*/false, a, /*neg_b=*/false, b,
      /*c_mod=*/(short)0, c, /*reuse_a=*/false, /*reuse_b=*/false);
}

__device__ __forceinline__ float sigmoidf_fast(float x) {
  return 1.0f / (1.0f + __expf(-x));
}

// ---------------------------------------------------------------------------
// One 16-byte async global->LDS transfer (gfx1250 ASYNCcnt path), with a
// synchronous load/store fallback if the builtin is unavailable.
// ---------------------------------------------------------------------------
__device__ __forceinline__ void async_copy16(const bf16* src, bf16* dst_lds) {
#if defined(__AMDGCN__) && __has_builtin(__builtin_amdgcn_global_load_async_to_lds_b128)
  __builtin_amdgcn_global_load_async_to_lds_b128(
      (__attribute__((address_space(1))) v4i*)src,
      (__attribute__((address_space(3))) v4i*)dst_lds, 0, 0);
#else
  *(v8bf*)dst_lds = *(const v8bf*)src;
#endif
}

__device__ __forceinline__ void wait_async_lds() {
#if defined(__AMDGCN__) && __has_builtin(__builtin_amdgcn_global_load_async_to_lds_b128)
#if __has_builtin(__builtin_amdgcn_s_wait_asynccnt)
  __builtin_amdgcn_s_wait_asynccnt(0);
#else
  asm volatile("s_wait_asynccnt 0" ::: "memory");
#endif
#endif
}

// ---------------------------------------------------------------------------
// Async global->LDS staging of one weight slice. gw is the bf16 weight matrix
// [4H x K] row-major; we stage rows {gate*kH + n0 + c : gate<4, c<16} into
// sdst[64][K]. One 16-byte chunk per iteration per thread.
// ---------------------------------------------------------------------------
template <int K>
__device__ __forceinline__ void stage_weight(const bf16* __restrict__ gw,
                                             bf16* sdst, int n0, int tid) {
  constexpr int chunksPerRow = K / 8;            // 8 bf16 per 16B chunk
  constexpr int total = 64 * chunksPerRow;
#pragma unroll 4
  for (int i = tid; i < total; i += 256) {
    const int r  = i / chunksPerRow;             // 0..63  (gate*16 + col)
    const int ck = i % chunksPerRow;
    const bf16* src = gw + (size_t)((r >> 4) * kH + n0 + (r & 15)) * K + ck * 8;
    bf16* dst = sdst + r * K + ck * 8;
    async_copy16(src, dst);
  }
}

// ---------------------------------------------------------------------------
// Device-scope epoch barrier across the 32 persistent workgroups.
// ctl[0]=arrive counter, ctl[1]=generation. Zeroed each launch by prep kernel.
// ---------------------------------------------------------------------------
__device__ __forceinline__ void grid_barrier(int* ctl, int* ep) {
  __syncthreads();
  if (threadIdx.x == 0) {
    __threadfence();
    const int e = ++(*ep);
    if (__hip_atomic_fetch_add(&ctl[0], 1, __ATOMIC_ACQ_REL,
                               __HIP_MEMORY_SCOPE_AGENT) == NWG - 1) {
      __hip_atomic_store(&ctl[0], 0, __ATOMIC_RELAXED, __HIP_MEMORY_SCOPE_AGENT);
      __hip_atomic_fetch_add(&ctl[1], 1, __ATOMIC_RELEASE, __HIP_MEMORY_SCOPE_AGENT);
    } else {
      while (__hip_atomic_load(&ctl[1], __ATOMIC_ACQUIRE,
                               __HIP_MEMORY_SCOPE_AGENT) < e) {
        __builtin_amdgcn_s_sleep(2);
      }
    }
    __threadfence();
  }
  __syncthreads();
}

// ---------------------------------------------------------------------------
// LSTM pointwise cell on one 16x16 gate tile quad (f32 state, bf16 h output).
// ---------------------------------------------------------------------------
__device__ __forceinline__ void lstm_cell_tile(v8f gi, v8f gf, v8f gg, v8f go,
                                               const float* __restrict__ bsum,
                                               float* __restrict__ cst,
                                               bf16* __restrict__ hout,
                                               int m0, int n0, int lane) {
  const int n  = n0 + (lane & 15);
  const float bi = bsum[n];
  const float bf_ = bsum[n + kH];
  const float bg = bsum[n + 2 * kH];
  const float bo = bsum[n + 3 * kH];
  const int mrow = m0 + ((lane >> 4) << 3);
#pragma unroll
  for (int r = 0; r < 8; ++r) {
    const int idx = (mrow + r) * kH + n;
    const float cprev = cst[idx];
    const float iv = sigmoidf_fast(gi[r] + bi);
    const float fv = sigmoidf_fast(gf[r] + bf_);
    const float gv = tanhf(gg[r] + bg);
    const float ov = sigmoidf_fast(go[r] + bo);
    const float cn = fv * cprev + iv * gv;
    cst[idx]  = cn;
    hout[idx] = (bf16)(ov * tanhf(cn));
  }
}

// ---------------------------------------------------------------------------
// Prep kernels
// ---------------------------------------------------------------------------
__global__ void k_zero_u32(unsigned int* __restrict__ p, int n) {
  for (int i = blockIdx.x * blockDim.x + threadIdx.x; i < n;
       i += gridDim.x * blockDim.x)
    p[i] = 0u;
}

__global__ void k_cvt_bf16(const float* __restrict__ s, bf16* __restrict__ d, int n) {
  for (int i = blockIdx.x * blockDim.x + threadIdx.x; i < n;
       i += gridDim.x * blockDim.x)
    d[i] = (bf16)s[i];
}

__global__ void k_bias_sum(const float* __restrict__ a, const float* __restrict__ b,
                           float* __restrict__ o, int n) {
  int i = blockIdx.x * blockDim.x + threadIdx.x;
  if (i < n) o[i] = a[i] + b[i];
}

// ---------------------------------------------------------------------------
// Input projection: xp[t][b][:] = x[b][t][:] @ proj_w^T + proj_b  (bf16 out)
// One 16x16 tile per wave; M = B*T = 65536 flat rows, N = D = 256.
// ---------------------------------------------------------------------------
__global__ __launch_bounds__(256) void k_proj(const float* __restrict__ x,
                                              const bf16* __restrict__ pw,
                                              const float* __restrict__ pb,
                                              bf16* __restrict__ xp) {
  const int lane = threadIdx.x & 31;
  const int wid  = (blockIdx.x * blockDim.x + threadIdx.x) >> 5;  // 0..65535
  const int nb = wid & 15;        // 16 N blocks
  const int mb = wid >> 4;        // 4096 M blocks
  const int m0 = mb * 16, n0 = nb * 16;

  v8f acc = {};
#pragma unroll 2
  for (int k = 0; k < kD; k += 32) {
    v16bf a = load_frag_f32(x, kD, m0, k, lane);       // x as [B*T, D]
    v16bf b = load_frag_bf16(pw, kD, n0, k, lane);     // B = proj_w^T
    acc = wmma_bf16(a, b, acc);
  }
  const int n = n0 + (lane & 15);
  const float bias = pb[n];
#pragma unroll
  for (int r = 0; r < 8; ++r) {
    const int row = m0 + r + ((lane >> 4) << 3);       // flat row = b*T + t
    const int bi = row >> 9;                           // / T
    const int ti = row & (kT - 1);                     // % T
    xp[((size_t)ti * kB + bi) * kD + n] = (bf16)(acc[r] + bias);
  }
}

// ---------------------------------------------------------------------------
// Persistent 512-step scan. Block = h-block (hb, 0..31); wave = batch block
// (mb, 0..7). All waves in a block share the SAME weight columns, which live
// in LDS for the whole kernel (224 KB -> 1 WG per WGP).
// ---------------------------------------------------------------------------
__global__ __launch_bounds__(256) void k_lstm_scan(
    const bf16* __restrict__ xp,
    const bf16* __restrict__ wx0, const bf16* __restrict__ wh0,
    const bf16* __restrict__ wx1, const bf16* __restrict__ wh1,
    const float* __restrict__ b0sum, const float* __restrict__ b1sum,
    float* __restrict__ c0, float* __restrict__ c1,
    bf16* __restrict__ h0buf, bf16* __restrict__ h1buf,
    int* ctl) {
  __shared__ alignas(16) bf16 swx0[64 * kD];   //  32 KB
  __shared__ alignas(16) bf16 swh0[64 * kH];   //  64 KB
  __shared__ alignas(16) bf16 swx1[64 * kH];   //  64 KB
  __shared__ alignas(16) bf16 swh1[64 * kH];   //  64 KB  -> 224 KB total

  const int lane = threadIdx.x & 31;
  const int hb = blockIdx.x;                   // h block 0..31
  const int mb = threadIdx.x >> 5;             // batch block 0..7
  const int m0 = mb * 16;
  const int n0 = hb * 16;

  // ---- one-time weight staging into LDS (reused by all 512 steps) ----
  stage_weight<kD>(wx0, swx0, n0, threadIdx.x);
  stage_weight<kH>(wh0, swh0, n0, threadIdx.x);
  stage_weight<kH>(wx1, swx1, n0, threadIdx.x);
  stage_weight<kH>(wh1, swh1, n0, threadIdx.x);
  wait_async_lds();
  __syncthreads();

  int epoch = 0;
  for (int t = 0; t < kT; ++t) {
    const bf16* h0r = h0buf + (size_t)(t & 1) * kBH;
    bf16*       h0w = h0buf + (size_t)((t + 1) & 1) * kBH;
    const bf16* h1r = h1buf + (size_t)(t & 1) * kBH;
    bf16*       h1w = h1buf + (size_t)((t + 1) & 1) * kBH;
    const bf16* xpt = xp + (size_t)t * kB * kD;
    if (t + 1 < kT)
      __builtin_prefetch(xp + (size_t)(t + 1) * kB * kD + m0 * kD, 0, 1);

    // -------- layer 0: gates = xp_t @ wx0^T + h0 @ wh0^T --------
    v8f gi = {}, gf = {}, gg = {}, go = {};
#pragma unroll 2
    for (int k = 0; k < kD; k += 32) {
      v16bf a = load_frag_bf16(xpt, kD, m0, k, lane);
      gi = wmma_bf16(a, load_frag_lds(swx0, kD, 0, k, lane), gi);
      gf = wmma_bf16(a, load_frag_lds(swx0, kD, 1, k, lane), gf);
      gg = wmma_bf16(a, load_frag_lds(swx0, kD, 2, k, lane), gg);
      go = wmma_bf16(a, load_frag_lds(swx0, kD, 3, k, lane), go);
    }
#pragma unroll 2
    for (int k = 0; k < kH; k += 32) {
      v16bf a = load_frag_bf16(h0r, kH, m0, k, lane);
      gi = wmma_bf16(a, load_frag_lds(swh0, kH, 0, k, lane), gi);
      gf = wmma_bf16(a, load_frag_lds(swh0, kH, 1, k, lane), gf);
      gg = wmma_bf16(a, load_frag_lds(swh0, kH, 2, k, lane), gg);
      go = wmma_bf16(a, load_frag_lds(swh0, kH, 3, k, lane), go);
    }
    lstm_cell_tile(gi, gf, gg, go, b0sum, c0, h0w, m0, n0, lane);

    grid_barrier(ctl, &epoch);   // publish h0w to layer-1 readers

    // -------- layer 1: gates = h0_new @ wx1^T + h1 @ wh1^T --------
    gi = (v8f){}; gf = (v8f){}; gg = (v8f){}; go = (v8f){};
#pragma unroll 2
    for (int k = 0; k < kH; k += 32) {
      v16bf a = load_frag_bf16(h0w, kH, m0, k, lane);
      gi = wmma_bf16(a, load_frag_lds(swx1, kH, 0, k, lane), gi);
      gf = wmma_bf16(a, load_frag_lds(swx1, kH, 1, k, lane), gf);
      gg = wmma_bf16(a, load_frag_lds(swx1, kH, 2, k, lane), gg);
      go = wmma_bf16(a, load_frag_lds(swx1, kH, 3, k, lane), go);
    }
#pragma unroll 2
    for (int k = 0; k < kH; k += 32) {
      v16bf a = load_frag_bf16(h1r, kH, m0, k, lane);
      gi = wmma_bf16(a, load_frag_lds(swh1, kH, 0, k, lane), gi);
      gf = wmma_bf16(a, load_frag_lds(swh1, kH, 1, k, lane), gf);
      gg = wmma_bf16(a, load_frag_lds(swh1, kH, 2, k, lane), gg);
      go = wmma_bf16(a, load_frag_lds(swh1, kH, 3, k, lane), go);
    }
    lstm_cell_tile(gi, gf, gg, go, b1sum, c1, h1w, m0, n0, lane);

    grid_barrier(ctl, &epoch);   // publish h1w / h0w before next step
  }
}

// ---------------------------------------------------------------------------
// Tiny head: out[b] = relu(h1 @ fc1^T + b1) @ fc2^T + b2
// ---------------------------------------------------------------------------
__global__ void k_fc_head(const bf16* __restrict__ h1,   // final buffer, [B,H]
                          const float* __restrict__ fw1, const float* __restrict__ fb1,
                          const float* __restrict__ fw2, const float* __restrict__ fb2,
                          float* __restrict__ out) {
  const int b = threadIdx.x;
  if (b >= kB) return;
  float o = 0.f;
  for (int j = 0; j < 32; ++j) {
    float acc = fb1[j];
    const float* w = fw1 + j * kH;
    const bf16* hr = h1 + b * kH;
    for (int k = 0; k < kH; ++k) acc = __fmaf_rn((float)hr[k], w[k], acc);
    acc = fmaxf(acc, 0.f);
    o = __fmaf_rn(acc, fw2[j], o);
  }
  out[b] = o + fb2[0];
}

// ---------------------------------------------------------------------------
// Launcher
// ---------------------------------------------------------------------------
extern "C" void kernel_launch(void* const* d_in, const int* in_sizes, int n_in,
                              void* d_out, int out_size, void* d_ws, size_t ws_size,
                              hipStream_t stream) {
  (void)in_sizes; (void)n_in; (void)out_size; (void)ws_size;

  const float* x      = (const float*)d_in[0];
  const float* proj_w = (const float*)d_in[1];
  const float* proj_b = (const float*)d_in[2];
  const float* wx0    = (const float*)d_in[3];
  const float* bx0    = (const float*)d_in[4];
  const float* wh0    = (const float*)d_in[5];
  const float* bh0    = (const float*)d_in[6];
  const float* wx1    = (const float*)d_in[7];
  const float* bx1    = (const float*)d_in[8];
  const float* wh1    = (const float*)d_in[9];
  const float* bh1    = (const float*)d_in[10];
  const float* fc1_w  = (const float*)d_in[11];
  const float* fc1_b  = (const float*)d_in[12];
  const float* fc2_w  = (const float*)d_in[13];
  const float* fc2_b  = (const float*)d_in[14];
  float* out = (float*)d_out;

  // ---- workspace carve-up (bytes) ----
  char* ws = (char*)d_ws;
  size_t off = 0;
  auto take = [&](size_t bytes) { char* p = ws + off; off += (bytes + 255) & ~size_t(255); return p; };
  int*   ctl    = (int*)  take(256);                    // barrier state
  float* c0     = (float*)take(kBH * 4);
  float* c1     = (float*)take(kBH * 4);
  bf16*  h0buf  = (bf16*) take(2 * kBH * 2);            // double-buffered
  bf16*  h1buf  = (bf16*) take(2 * kBH * 2);
  float* b0sum  = (float*)take(kG * 4);
  float* b1sum  = (float*)take(kG * 4);
  bf16*  pw_bf  = (bf16*) take((size_t)kD * kD * 2);
  bf16*  wx0_bf = (bf16*) take((size_t)kG * kD * 2);
  bf16*  wh0_bf = (bf16*) take((size_t)kG * kH * 2);
  bf16*  wx1_bf = (bf16*) take((size_t)kG * kH * 2);
  bf16*  wh1_bf = (bf16*) take((size_t)kG * kH * 2);
  bf16*  xp_bf  = (bf16*) take((size_t)kT * kB * kD * 2);

  // 1) zero barrier state + c0/c1 + both h buffers (contiguous from ws start)
  {
    const int nwords = (int)((256 + 2 * kBH * 4 + 2 * (2 * kBH * 2)) / 4);
    k_zero_u32<<<(nwords + 255) / 256, 256, 0, stream>>>((unsigned int*)ws, nwords);
  }
  // 2) weights -> bf16
  k_cvt_bf16<<<(kD * kD + 255) / 256, 256, 0, stream>>>(proj_w, pw_bf, kD * kD);
  k_cvt_bf16<<<(kG * kD + 255) / 256, 256, 0, stream>>>(wx0, wx0_bf, kG * kD);
  k_cvt_bf16<<<(kG * kH + 255) / 256, 256, 0, stream>>>(wh0, wh0_bf, kG * kH);
  k_cvt_bf16<<<(kG * kH + 255) / 256, 256, 0, stream>>>(wx1, wx1_bf, kG * kH);
  k_cvt_bf16<<<(kG * kH + 255) / 256, 256, 0, stream>>>(wh1, wh1_bf, kG * kH);
  // 3) combined gate biases
  k_bias_sum<<<(kG + 255) / 256, 256, 0, stream>>>(bx0, bh0, b0sum, kG);
  k_bias_sum<<<(kG + 255) / 256, 256, 0, stream>>>(bx1, bh1, b1sum, kG);

  // 4) input projection -> xp (bf16, time-major)
  k_proj<<<(kB * kT / 16) * (kD / 16) / 8, 256, 0, stream>>>(x, pw_bf, proj_b, xp_bf);

  // 5) persistent 512-step scan (one block per h-block)
  k_lstm_scan<<<NWG, 256, 0, stream>>>(xp_bf, wx0_bf, wh0_bf, wx1_bf, wh1_bf,
                                       b0sum, b1sum, c0, c1, h0buf, h1buf, ctl);

  // 6) head: final h1 lives in buffer 0 since T is even
  k_fc_head<<<1, 128, 0, stream>>>(h1buf, fc1_w, fc1_b, fc2_w, fc2_b, out);
}
